// Mamba2Layer_22849226015240
// MI455X (gfx1250) — matrix operand order repaired and checked by reading, not compiled, not run on hardware
//
#include <hip/hip_runtime.h>
#include <math.h>

// ---------------- Mamba2 layer constants ----------------
#define SEQ_LEN   4096
#define HIDDEN    2048
#define INTER     4096
#define STATE     128
#define NHEADS    64
#define HEAD_DIM  64
#define KCONV     4
#define CHUNK     256
#define NCHUNK    (SEQ_LEN / CHUNK)            // 16
#define CONV_DIM  (INTER + 2 * STATE)          // 4352
#define IN_DIM    (CONV_DIM + NHEADS)          // 4416

// ---------------- WMMA types / helpers (gfx1250 wave32) ----------------
typedef __attribute__((ext_vector_type(16))) _Float16 v16h;
typedef __attribute__((ext_vector_type(8)))  _Float16 v8h;
typedef __attribute__((ext_vector_type(8)))  float    v8f;
typedef __attribute__((ext_vector_type(4)))  unsigned int u32x4;
typedef __attribute__((ext_vector_type(4)))  int i32x4;
typedef __attribute__((ext_vector_type(8)))  int i32x8;

#if __has_builtin(__builtin_amdgcn_tensor_load_to_lds) && \
    __has_builtin(__builtin_amdgcn_s_wait_tensorcnt)
#define HAVE_TDM 1
#else
#define HAVE_TDM 0
#endif

static __device__ __forceinline__ v8f wmma16(v16h a, v16h b, v8f c) {
  // D(16x16,f32) = A(16x32,f16) * B(32x16,f16) + C  -> v_wmma_f32_16x16x32_f16
  return __builtin_amdgcn_wmma_f32_16x16x32_f16(false, a, false, b, (short)0, c,
                                                false, false);
}

// A fragment: 16(M) x 32(K) f16 from LDS [row][k], stride ld (halves).
static __device__ __forceinline__ v16h frag_a(const _Float16* S, int row0,
                                              int k0, int ld, int lane) {
  const _Float16* p = S + (row0 + (lane & 15)) * ld + k0 + ((lane >> 4) << 3);
  v8h lo = *(const v8h*)p;
  v8h hi = *(const v8h*)(p + 16);
  return __builtin_shufflevector(lo, hi, 0, 1, 2, 3, 4, 5, 6, 7, 8, 9, 10, 11,
                                 12, 13, 14, 15);
}

// B fragment: 32(K) x 16(N) f16 from LDS transposed [col][k], stride ld.
static __device__ __forceinline__ v16h frag_b(const _Float16* S, int col0,
                                              int k0, int ld, int lane) {
  const _Float16* p = S + (col0 + (lane & 15)) * ld + k0 + ((lane >> 4) << 3);
  v8h lo = *(const v8h*)p;
  v8h hi = *(const v8h*)(p + 16);
  return __builtin_shufflevector(lo, hi, 0, 1, 2, 3, 4, 5, 6, 7, 8, 9, 10, 11,
                                 12, 13, 14, 15);
}

#if HAVE_TDM
// Issue a TDM 2D tile load (fp32 elements) global -> LDS. Wave-level op,
// EXEC ignored; descriptor per CDNA5 ISA (D# groups 0/1).
static __device__ __forceinline__ void tdm_load_2d_f32(const float* gsrc,
                                                       void* lds_dst, int tile0,
                                                       int tile1, int tdim0,
                                                       int tdim1, int stride0) {
  unsigned long long ga = (unsigned long long)(size_t)gsrc;
  unsigned lds_off = (unsigned)(size_t)lds_dst;  // flat->LDS: low 32 bits
  u32x4 g0 = {1u,                                // count=1, user mode
              lds_off,                           // lds_addr
              (unsigned)(ga & 0xFFFFFFFFull),    // global_addr[31:0]
              (unsigned)((ga >> 32) & 0x01FFFFFFull) | 0x80000000u};  // type=2
  i32x8 g1 = {
      (int)(2u << 16),                                    // data_size = 4B
      (int)(((unsigned)tdim0 & 0xFFFFu) << 16),           // tensor_dim0 lo16
      (int)((((unsigned)tdim0 >> 16) & 0xFFFFu) |
            (((unsigned)tdim1 & 0xFFFFu) << 16)),         // t0 hi | t1 lo
      (int)((((unsigned)tdim1 >> 16) & 0xFFFFu) |
            (((unsigned)tile0 & 0xFFFFu) << 16)),         // t1 hi | tile_dim0
      (int)((unsigned)tile1 & 0xFFFFu),                   // tile_dim1, dim2=0
      (int)stride0,                                       // dim0 stride lo32
      0, 0};
  i32x4 z4 = {0, 0, 0, 0};
#if __clang_major__ >= 23
  i32x8 z8 = {0, 0, 0, 0, 0, 0, 0, 0};
  __builtin_amdgcn_tensor_load_to_lds(g0, g1, z4, z4, z8, 0);
#else
  __builtin_amdgcn_tensor_load_to_lds(g0, g1, z4, z4, 0);
#endif
}
#endif

// ---------------- K1/K9: NT GEMM, C[M,N] = A[M,K] * B[N,K]^T ------------------
// 128x128 block tile, K-step 32, 8 waves: each wave 32x64 output (8 wmma/step).
// TDM ping-pong stages fp32 K-tiles into LDS; waves convert to f16 and WMMA.
__global__ __launch_bounds__(256) void gemm_nt_wmma(
    const float* __restrict__ A, const float* __restrict__ B,
    float* __restrict__ C, int M, int N, int K) {
  extern __shared__ char smem[];
  float* stA0 = (float*)smem;                   // [128][32] fp32 stage, buf0
  float* stB0 = stA0 + 128 * 32;
  float* stA1 = stB0 + 128 * 32;                // buf1
  float* stB1 = stA1 + 128 * 32;
  _Float16* sA = (_Float16*)(stB1 + 128 * 32);  // [128][40] f16
  _Float16* sB = sA + 128 * 40;                 // [128][40] f16
  const int LD = 40;
  const int tid = threadIdx.x, lane = tid & 31, wave = tid >> 5;
  const int bm = blockIdx.y * 128, bn = blockIdx.x * 128;
  const int wrow = (wave & 3) * 32;             // wave row strip
  const int wcol = (wave >> 2) * 64;            // wave col strip
  const int lrow = tid >> 1;                    // loader row 0..127
  const int lc0 = (tid & 1) * 16;               // loader col group
  const int nsteps = K / 32;

  v8f acc[2][4] = {};

#if HAVE_TDM
  if (wave == 0) {  // prime the pipeline: tile 0 into buffer 0
    tdm_load_2d_f32(A + (size_t)bm * K, stA0, 32, 128, K, M, K);
    tdm_load_2d_f32(B + (size_t)bn * K, stB0, 32, 128, K, N, K);
  }
#endif

  for (int s = 0; s < nsteps; ++s) {
    const int k0 = s * 32;
#if HAVE_TDM
    float* stA = (s & 1) ? stA1 : stA0;
    float* stB = (s & 1) ? stB1 : stB0;
    if (wave == 0) __builtin_amdgcn_s_wait_tensorcnt(0);
    __syncthreads();  // tensor data landed + previous wmma LDS reads retired
    if (wave == 0 && s + 1 < nsteps) {  // async-prefetch next K-tile via TDM
      float* nA = (s & 1) ? stA0 : stA1;
      float* nB = (s & 1) ? stB0 : stB1;
      tdm_load_2d_f32(A + (size_t)bm * K + k0 + 32, nA, 32, 128, K, M, K);
      tdm_load_2d_f32(B + (size_t)bn * K + k0 + 32, nB, 32, 128, K, N, K);
    }
#pragma unroll
    for (int i = 0; i < 16; i++) {  // fp32 stage -> f16 operand tiles
      sA[lrow * LD + lc0 + i] = (_Float16)stA[lrow * 32 + lc0 + i];
      sB[lrow * LD + lc0 + i] = (_Float16)stB[lrow * 32 + lc0 + i];
    }
#else
    __syncthreads();
    const float* ga = A + (size_t)(bm + lrow) * K + k0 + lc0;
    const float* gb = B + (size_t)(bn + lrow) * K + k0 + lc0;
    const bool bok = (bn + lrow) < N;
    if (k0 + 32 < K) {
      __builtin_prefetch(ga + 32, 0, 1);
      if (bok) __builtin_prefetch(gb + 32, 0, 1);
    }
#pragma unroll
    for (int i = 0; i < 16; i++) {
      sA[lrow * LD + lc0 + i] = (_Float16)ga[i];
      sB[lrow * LD + lc0 + i] = bok ? (_Float16)gb[i] : (_Float16)0.0f;
    }
#endif
    __syncthreads();
    v16h a0 = frag_a(sA, wrow, 0, LD, lane);
    v16h a1 = frag_a(sA, wrow + 16, 0, LD, lane);
#pragma unroll
    for (int j = 0; j < 4; j++) {
      v16h b = frag_b(sB, wcol + j * 16, 0, LD, lane);
      acc[0][j] = wmma16(a0, b, acc[0][j]);
      acc[1][j] = wmma16(a1, b, acc[1][j]);
    }
  }
  const int colp = lane & 15, rb = (lane >> 4) ? 8 : 0;
#pragma unroll
  for (int r = 0; r < 2; r++)
#pragma unroll
    for (int j = 0; j < 4; j++) {
      int ccol = bn + wcol + j * 16 + colp;
      if (ccol < N) {
#pragma unroll
        for (int v = 0; v < 8; v++)
          C[(size_t)(bm + wrow + r * 16 + rb + v) * N + ccol] = acc[r][j][v];
      }
    }
}

// ---------------- K2: depthwise causal conv (k=4) + bias + SiLU ---------------
__global__ void conv_silu_kernel(const float* __restrict__ xbcdt,
                                 const float* __restrict__ cw,
                                 const float* __restrict__ cb,
                                 float* __restrict__ xbc) {
  size_t idx = (size_t)blockIdx.x * blockDim.x + threadIdx.x;
  const size_t total = (size_t)SEQ_LEN * CONV_DIM;
  const size_t stride = (size_t)gridDim.x * blockDim.x;
  for (; idx < total; idx += stride) {
    int c = (int)(idx % CONV_DIM);
    int l = (int)(idx / CONV_DIM);
    float acc = cb[c];
#pragma unroll
    for (int t = 0; t < KCONV; t++) {
      int ls = l - (KCONV - 1) + t;
      if (ls >= 0) acc += cw[c * KCONV + t] * xbcdt[(size_t)ls * IN_DIM + c];
    }
    xbc[idx] = acc / (1.0f + __expf(-acc));            // SiLU
  }
}

// ---------------- K3: dt = clip(softplus(raw+bias)), dA = -exp(A_log)*dt ------
__global__ void dt_prep_kernel(const float* __restrict__ xbcdt,
                               const float* __restrict__ dtb,
                               const float* __restrict__ Alog,
                               float* __restrict__ dtv,
                               float* __restrict__ dAv) {
  int idx = blockIdx.x * blockDim.x + threadIdx.x;
  if (idx >= SEQ_LEN * NHEADS) return;
  int h = idx & (NHEADS - 1), l = idx >> 6;
  float v = xbcdt[(size_t)l * IN_DIM + CONV_DIM + h] + dtb[h];
  float sp = (v > 20.f) ? v : log1pf(__expf(v));
  sp = fminf(fmaxf(sp, 0.0f), 100.0f);
  dtv[idx] = sp;
  dAv[idx] = -__expf(Alog[h]) * sp;
}

// ---------------- K4: per-(h,c) inclusive cumsum of dA over the chunk ---------
__global__ __launch_bounds__(256) void cumsum_kernel(
    const float* __restrict__ dAv, float* __restrict__ Acs,
    float* __restrict__ csum) {
  __shared__ float buf[2][CHUNK];
  int hc = blockIdx.x;                     // h*NCHUNK + c
  int h = hc >> 4, c = hc & (NCHUNK - 1);
  int tid = threadIdx.x;
  buf[0][tid] = dAv[((size_t)(c * CHUNK + tid)) * NHEADS + h];
  __syncthreads();
  int src = 0;
  for (int off = 1; off < CHUNK; off <<= 1) {
    float x = buf[src][tid];
    if (tid >= off) x += buf[src][tid - off];
    buf[src ^ 1][tid] = x;
    __syncthreads();
    src ^= 1;
  }
  float r = buf[src][tid];
  Acs[(size_t)hc * CHUNK + tid] = r;
  if (tid == CHUNK - 1) csum[hc] = r;
}

// ---------------- K5: states[c,h,p,n] = sum_l (x*dt*decay)[p,l] * B[l,n] ------
// WMMA:  M=64(p), N=128(n), K=256(l)
__global__ __launch_bounds__(256) void states_kernel(
    const float* __restrict__ xBC, const float* __restrict__ dtv,
    const float* __restrict__ Acs, float* __restrict__ states) {
  extern __shared__ char smem[];
  const int LDL = CHUNK + 8;                        // 264 halves
  _Float16* sXT = (_Float16*)smem;                  // [64][264]  (p, l)
  _Float16* sBT = sXT + HEAD_DIM * LDL;             // [128][264] (n, l)
  float* sAcs = (float*)(sBT + STATE * LDL);        // [256]
  const int c = blockIdx.x, h = blockIdx.y;
  const int tid = threadIdx.x, lane = tid & 31, wave = tid >> 5;
  const int lbase = c * CHUNK;

  sAcs[tid] = Acs[(size_t)(h * NCHUNK + c) * CHUNK + tid];
  __syncthreads();
  const float last = sAcs[CHUNK - 1];
  for (int i = tid; i < HEAD_DIM * CHUNK; i += 256) {
    int p = i >> 8, l = i & (CHUNK - 1);
    float w = __expf(last - sAcs[l]);
    float xv = xBC[(size_t)(lbase + l) * CONV_DIM + h * HEAD_DIM + p] *
               dtv[(size_t)(lbase + l) * NHEADS + h];
    sXT[p * LDL + l] = (_Float16)(xv * w);
  }
  for (int i = tid; i < STATE * CHUNK; i += 256) {
    int n = i >> 8, l = i & (CHUNK - 1);
    sBT[n * LDL + l] = (_Float16)xBC[(size_t)(lbase + l) * CONV_DIM + INTER + n];
  }
  __syncthreads();

  const int rt = (wave & 3) * 16;                   // p row tile
  const int ch = (wave >> 2) * 64;                  // n col group
  v8f acc[4] = {};
  for (int k0 = 0; k0 < CHUNK; k0 += 32) {
    v16h a = frag_a(sXT, rt, k0, LDL, lane);
#pragma unroll
    for (int j = 0; j < 4; j++) {
      v16h b = frag_b(sBT, ch + j * 16, k0, LDL, lane);
      acc[j] = wmma16(a, b, acc[j]);
    }
  }
  const int rb = rt + ((lane >> 4) ? 8 : 0), coln = lane & 15;
#pragma unroll
  for (int j = 0; j < 4; j++)
#pragma unroll
    for (int v = 0; v < 8; v++) {
      int p = rb + v, n = ch + j * 16 + coln;
      states[((((size_t)c * NHEADS + h) * HEAD_DIM + p) * STATE) + n] = acc[j][v];
    }
}

// ---------------- K6: sequential inter-chunk state recurrence -----------------
__global__ __launch_bounds__(256) void chunk_scan_kernel(
    const float* __restrict__ states, const float* __restrict__ csum,
    float* __restrict__ sprev) {
  const int h = blockIdx.x, tid = threadIdx.x;
  float prev[32];
#pragma unroll
  for (int k = 0; k < 32; k++) prev[k] = 0.0f;
  for (int c = 0; c < NCHUNK; c++) {
    if (c + 1 < NCHUNK)  // pull next chunk's lines toward L2 early
      __builtin_prefetch(states + (((size_t)c * NHEADS + h) * 8192) + tid, 0, 1);
    float e = (c == 0) ? 0.0f : __expf(csum[h * NCHUNK + c - 1]);
    for (int k = 0; k < 32; k++) {
      int idx = tid + k * 256;                         // p*128+n
      float s = (c == 0) ? 0.0f
                         : states[(((size_t)(c - 1) * NHEADS + h) * 8192) + idx];
      prev[k] = prev[k] * e + s;
      sprev[(((size_t)c * NHEADS + h) * 8192) + idx] = prev[k];
    }
  }
}

// ---------------- K7: Y = diag(e^Acs)*(C Sprev^T) + (CB^T ⊙ decaymask) X + Dx -
__global__ __launch_bounds__(256) void ssd_y_kernel(
    const float* __restrict__ xBC, const float* __restrict__ dtv,
    const float* __restrict__ Acs, const float* __restrict__ sprev,
    const float* __restrict__ Dp, float* __restrict__ Y) {
  extern __shared__ char smem[];
  const int LDN = STATE + 8;                        // 136 halves
  const int LDL = CHUNK + 8;                        // 264 halves
  _Float16* sC  = (_Float16*)smem;                  // [256][136] (l, n)
  _Float16* sB  = sC + CHUNK * LDN;                 // [256][136] (s, n)
  _Float16* sXT = sB + CHUNK * LDN;                 // [64][264]  (p, s)
  _Float16* sSp = sXT + HEAD_DIM * LDL;             // [64][136]  (p, n)
  float* sAcs   = (float*)(sSp + HEAD_DIM * LDN);   // [256]
  _Float16* sM  = (_Float16*)(sAcs + CHUNK);        // [8][32][40] per-wave
  const int c = blockIdx.x, h = blockIdx.y;
  const int tid = threadIdx.x, lane = tid & 31, wave = tid >> 5;
  const int lbase = c * CHUNK;

  for (int i = tid; i < CHUNK * STATE; i += 256) {
    int l = i >> 7, n = i & (STATE - 1);
    const float* row = xBC + (size_t)(lbase + l) * CONV_DIM;
    sB[l * LDN + n] = (_Float16)row[INTER + n];
    sC[l * LDN + n] = (_Float16)row[INTER + STATE + n];
  }
  for (int i = tid; i < HEAD_DIM * CHUNK; i += 256) {
    int p = i >> 8, s = i & (CHUNK - 1);
    float xv = xBC[(size_t)(lbase + s) * CONV_DIM + h * HEAD_DIM + p];
    sXT[p * LDL + s] = (_Float16)(xv * dtv[(size_t)(lbase + s) * NHEADS + h]);
  }
  for (int i = tid; i < HEAD_DIM * STATE; i += 256) {
    int p = i >> 7, n = i & (STATE - 1);
    sSp[p * LDN + n] =
        (_Float16)sprev[((((size_t)c * NHEADS + h) * HEAD_DIM + p) * STATE) + n];
  }
  sAcs[tid] = Acs[(size_t)(h * NCHUNK + c) * CHUNK + tid];
  __syncthreads();

  const int l0 = wave * 32;                          // 32 output rows per wave
  _Float16* myM = sM + wave * 32 * 40;
  v8f acc[2][4] = {};

  // ---- Y_off = C @ Sprev^T  (K = n = 128) ----
#pragma unroll
  for (int k0 = 0; k0 < STATE; k0 += 32) {
    v16h a0 = frag_a(sC, l0, k0, LDN, lane);
    v16h a1 = frag_a(sC, l0 + 16, k0, LDN, lane);
#pragma unroll
    for (int j = 0; j < 4; j++) {
      v16h b = frag_b(sSp, j * 16, k0, LDN, lane);
      acc[0][j] = wmma16(a0, b, acc[0][j]);
      acc[1][j] = wmma16(a1, b, acc[1][j]);
    }
  }
  {  // row scale by exp(Acs[l]) for the off-diagonal contribution
    const int rb0 = l0 + ((lane >> 4) ? 8 : 0);
#pragma unroll
    for (int r = 0; r < 2; r++)
#pragma unroll
      for (int v = 0; v < 8; v++) {
        float e = __expf(sAcs[rb0 + r * 16 + v]);
#pragma unroll
        for (int j = 0; j < 4; j++) acc[r][j][v] *= e;
      }
  }

  // ---- Y_diag: per 32-wide s tile: G = C B^T, mask+decay, then M @ X ----
  for (int st = 0; st < 8; ++st) {
    const int s0 = st * 32;
    v8f g[2][2] = {};
#pragma unroll
    for (int k0 = 0; k0 < STATE; k0 += 32) {
      v16h a0 = frag_a(sC, l0, k0, LDN, lane);
      v16h a1 = frag_a(sC, l0 + 16, k0, LDN, lane);
      v16h b0 = frag_b(sB, s0, k0, LDN, lane);
      v16h b1 = frag_b(sB, s0 + 16, k0, LDN, lane);
      g[0][0] = wmma16(a0, b0, g[0][0]);
      g[0][1] = wmma16(a0, b1, g[0][1]);
      g[1][0] = wmma16(a1, b0, g[1][0]);
      g[1][1] = wmma16(a1, b1, g[1][1]);
    }
    const int col = lane & 15, rb = (lane >> 4) ? 8 : 0;
#pragma unroll
    for (int r = 0; r < 2; r++)
#pragma unroll
      for (int j = 0; j < 2; j++)
#pragma unroll
        for (int v = 0; v < 8; v++) {
          int row = l0 + r * 16 + rb + v;            // l within chunk
          int sc = s0 + j * 16 + col;                // s within chunk
          float gv = g[r][j][v];
          gv = (sc <= row) ? gv * __expf(sAcs[row] - sAcs[sc]) : 0.0f;
          myM[(r * 16 + rb + v) * 40 + j * 16 + col] = (_Float16)gv;
        }
    v16h ma0 = frag_a(myM, 0, 0, 40, lane);          // per-wave scratch
    v16h ma1 = frag_a(myM, 16, 0, 40, lane);
#pragma unroll
    for (int j = 0; j < 4; j++) {
      v16h b = frag_b(sXT, j * 16, s0, LDL, lane);
      acc[0][j] = wmma16(ma0, b, acc[0][j]);
      acc[1][j] = wmma16(ma1, b, acc[1][j]);
    }
  }

  // ---- + D[h] * x_ssm, store Y[l, h*64+p] ----
  const float Dh = Dp[h];
  const int colp = lane & 15, rb = (lane >> 4) ? 8 : 0;
#pragma unroll
  for (int r = 0; r < 2; r++)
#pragma unroll
    for (int j = 0; j < 4; j++)
#pragma unroll
      for (int v = 0; v < 8; v++) {
        int l = l0 + r * 16 + rb + v;
        int p = j * 16 + colp;
        size_t gl = (size_t)(lbase + l);
        float xv = xBC[gl * CONV_DIM + h * HEAD_DIM + p];
        Y[gl * INTER + h * HEAD_DIM + p] = acc[r][j][v] + Dh * xv;
      }
}

// ---------------- K8: RMS-style norm over INTER ------------------------------
__global__ __launch_bounds__(256) void rmsnorm_kernel(float* __restrict__ y,
                                                      const float* __restrict__ w) {
  __shared__ float red[256];
  const int l = blockIdx.x, tid = threadIdx.x;
  float* row = y + (size_t)l * INTER;
  float s = 0.0f;
  for (int i = tid; i < INTER; i += 256) {
    float v = row[i];
    s += v * v;
  }
  red[tid] = s;
  __syncthreads();
  for (int off = 128; off > 0; off >>= 1) {
    if (tid < off) red[tid] += red[tid + off];
    __syncthreads();
  }
  float inv = rsqrtf(red[0] / (float)INTER + 1e-6f);
  for (int i = tid; i < INTER; i += 256) row[i] = row[i] * inv * w[i];
}

// ---------------- workspace layout (bytes) -----------------------------------
static const size_t OFF_XBCDT  = 0;                                   // 4096*4416*4
static const size_t OFF_XBC    = (size_t)SEQ_LEN * IN_DIM * 4;
static const size_t OFF_DT     = OFF_XBC + (size_t)SEQ_LEN * CONV_DIM * 4;
static const size_t OFF_DA     = OFF_DT + (size_t)SEQ_LEN * NHEADS * 4;
static const size_t OFF_ACS    = OFF_DA + (size_t)SEQ_LEN * NHEADS * 4;
static const size_t OFF_CSUM   = OFF_ACS + (size_t)NHEADS * NCHUNK * CHUNK * 4;
static const size_t OFF_STATES = OFF_CSUM + (size_t)NHEADS * NCHUNK * 4;
static const size_t OFF_SPREV  =
    OFF_STATES + (size_t)NCHUNK * NHEADS * HEAD_DIM * STATE * 4;
// y (67MB) aliases the xBC_dt region (72MB), which is dead after dt_prep.

extern "C" void kernel_launch(void* const* d_in, const int* in_sizes, int n_in,
                              void* d_out, int out_size, void* d_ws,
                              size_t ws_size, hipStream_t stream) {
  (void)in_sizes; (void)n_in; (void)out_size; (void)ws_size;
  const float* x      = (const float*)d_in[0];
  const float* W_in   = (const float*)d_in[1];
  const float* conv_w = (const float*)d_in[2];
  const float* conv_b = (const float*)d_in[3];
  const float* dt_b   = (const float*)d_in[4];
  const float* A_log  = (const float*)d_in[5];
  const float* Dv     = (const float*)d_in[6];
  const float* norm_w = (const float*)d_in[7];
  const float* W_out  = (const float*)d_in[8];

  char* ws = (char*)d_ws;
  float* xbcdt  = (float*)(ws + OFF_XBCDT);
  float* y      = (float*)(ws + OFF_XBCDT);   // alias
  float* xbc    = (float*)(ws + OFF_XBC);
  float* dtv    = (float*)(ws + OFF_DT);
  float* dAv    = (float*)(ws + OFF_DA);
  float* acs    = (float*)(ws + OFF_ACS);
  float* csum   = (float*)(ws + OFF_CSUM);
  float* states = (float*)(ws + OFF_STATES);
  float* sprev  = (float*)(ws + OFF_SPREV);

  // stage (4 x 128x32 fp32) + 2 x 128x40 f16 tiles = 86016 B (fixed layout
  // regardless of TDM availability so host/device sizes agree)
  const size_t sh_gemm = (size_t)4 * 128 * 32 * sizeof(float) +
                         (size_t)2 * 128 * 40 * sizeof(_Float16);
  const size_t sh_states =
      ((size_t)HEAD_DIM * 264 + (size_t)STATE * 264) * sizeof(_Float16) +
      CHUNK * sizeof(float);
  const size_t sh_ssd =
      ((size_t)2 * CHUNK * 136 + (size_t)HEAD_DIM * 264 +
       (size_t)HEAD_DIM * 136 + (size_t)8 * 32 * 40) * sizeof(_Float16) +
      CHUNK * sizeof(float);

  // 1) xBC_dt = x @ W_in^T            [4096,4416]  (N edge handled by TDM OOB)
  gemm_nt_wmma<<<dim3((IN_DIM + 127) / 128, SEQ_LEN / 128), 256, sh_gemm,
                 stream>>>(x, W_in, xbcdt, SEQ_LEN, IN_DIM, HIDDEN);
  // 2) depthwise conv + SiLU -> xBC   [4096,4352]
  conv_silu_kernel<<<4096, 256, 0, stream>>>(xbcdt, conv_w, conv_b, xbc);
  // 3) dt / dA                        [4096,64]
  dt_prep_kernel<<<(SEQ_LEN * NHEADS) / 256, 256, 0, stream>>>(
      xbcdt, dt_b, A_log, dtv, dAv);
  // 4) per-chunk cumsum of dA
  cumsum_kernel<<<NHEADS * NCHUNK, 256, 0, stream>>>(dAv, acs, csum);
  // 5) intra-chunk states (WMMA)
  states_kernel<<<dim3(NCHUNK, NHEADS), 256, sh_states, stream>>>(
      xbc, dtv, acs, states);
  // 6) inter-chunk recurrence
  chunk_scan_kernel<<<NHEADS, 256, 0, stream>>>(states, csum, sprev);
  // 7) Y = Y_diag + Y_off + D*x (WMMA)
  ssd_y_kernel<<<dim3(NCHUNK, NHEADS), 256, sh_ssd, stream>>>(
      xbc, dtv, acs, sprev, Dv, y);
  // 8) RMS norm (in place on y)
  rmsnorm_kernel<<<SEQ_LEN, 256, 0, stream>>>(y, norm_w);
  // 9) out = y @ W_out^T              [4096,2048]
  gemm_nt_wmma<<<dim3(HIDDEN / 128, SEQ_LEN / 128), 256, sh_gemm, stream>>>(
      y, W_out, (float*)d_out, SEQ_LEN, HIDDEN, INTER);
}
